// CellWiseLSTM_75084618269312
// MI455X (gfx1250) — matrix operand order repaired and checked by reading, not compiled, not run on hardware
//
#include <hip/hip_runtime.h>
#include <math.h>

#define BATCH 32
#define SEQ 512
#define HID 1024
#define NLAYER 2
#define GATES (4 * HID)   // 4096
#define KDIM  (2 * HID)   // 2048 (x concat h)
#define KCHUNK 512        // K staged per LDS chunk
#define RS (KCHUNK + 8)   // padded LDS row stride; matches TDM pad 4 DWORDs / 256 DWORDs
#define NBLK 32           // persistent workgroups (= GATES / 128 columns per WG)

typedef __attribute__((ext_vector_type(16))) __bf16 v16bf;
typedef __attribute__((ext_vector_type(8)))  float  v8f;
typedef __attribute__((ext_vector_type(4)))  unsigned int u32x4;
typedef __attribute__((ext_vector_type(8)))  int i32x8;
typedef __attribute__((ext_vector_type(4)))  int i32x4;

union V16 { u32x4 u[2]; v16bf v; };

__device__ __forceinline__ float sigmoidf_(float x) { return 1.0f / (1.0f + __expf(-x)); }

// ---------------- prep kernels ----------------

__global__ void prep_w(const float* __restrict__ Wi, const float* __restrict__ Wh,
                       const float* __restrict__ bi, const float* __restrict__ bh,
                       __bf16* __restrict__ Wcat, float* __restrict__ bias, int total) {
  int idx = blockIdx.x * blockDim.x + threadIdx.x;
  if (idx >= total) return;
  int l = idx / (GATES * KDIM);
  int rem = idx % (GATES * KDIM);
  int nrow = rem / KDIM;
  int k = rem % KDIM;
  float w = (k < HID) ? Wi[(size_t)l * GATES * HID + (size_t)nrow * HID + k]
                      : Wh[(size_t)l * GATES * HID + (size_t)nrow * HID + (k - HID)];
  Wcat[idx] = (__bf16)w;
  if (k == 0) bias[l * GATES + nrow] = bi[l * GATES + nrow] + bh[l * GATES + nrow];
}

__global__ void prep_x(const float* __restrict__ x, __bf16* __restrict__ xb, int total) {
  int idx = blockIdx.x * blockDim.x + threadIdx.x;
  if (idx >= total) return;
  int s = idx / (BATCH * HID);
  int rem = idx % (BATCH * HID);
  int b = rem / HID;
  int h = rem % HID;
  xb[idx] = (__bf16)x[(size_t)b * SEQ * HID + (size_t)s * HID + h];
}

__global__ void prep_state(const float* __restrict__ h0, const float* __restrict__ c0,
                           __bf16* __restrict__ h0b, float* __restrict__ cb,
                           unsigned* __restrict__ bar, int total) {
  int idx = blockIdx.x * blockDim.x + threadIdx.x;
  if (idx >= total) return;
  h0b[idx] = (__bf16)h0[idx];
  cb[idx] = c0[idx];
  if (idx == 0) *bar = 0u;   // reset grid barrier every call (determinism)
}

// ---------------- device-wide barrier (32 co-resident WGs) ----------------
__device__ __forceinline__ void grid_sync(unsigned* bar, unsigned target) {
  __syncthreads();
  if (threadIdx.x == 0) {
    __threadfence();  // release: publish h/c stores to device scope (L2)
    __hip_atomic_fetch_add(bar, 1u, __ATOMIC_RELEASE, __HIP_MEMORY_SCOPE_AGENT);
    while (__hip_atomic_load(bar, __ATOMIC_ACQUIRE, __HIP_MEMORY_SCOPE_AGENT) < target) {
      __builtin_amdgcn_s_sleep(1);
    }
    __threadfence();  // acquire side
  }
  __syncthreads();
}

// ---------------- persistent LSTM kernel: all layers, all timesteps ----------------
// Exactly NBLK=32 WGs on a >=32-WGP device -> 1 WG per WGP; occupancy 1 is intended,
// so give the compiler the full VGPR budget (no spills of the weight stream).
__global__ __launch_bounds__(256, 1) void lstm_persistent(
    const __bf16* __restrict__ Wcat, const float* __restrict__ biasAll,
    const __bf16* __restrict__ xb, __bf16* __restrict__ hall,
    const __bf16* __restrict__ h0b, float* __restrict__ cbAll,
    float* __restrict__ out, float* __restrict__ hTall, float* __restrict__ cTall,
    unsigned* __restrict__ bar) {
  __shared__ __bf16 xh[BATCH][RS];       // TDM-staged activation chunk (padded rows)
  __shared__ float gbuf[4][BATCH][32];   // gate pre-activations for this WG's columns

  const int tid = threadIdx.x;
  const int lane = tid & 31;
  const int wave = tid >> 5;
  const int gate = wave >> 1;
  const int jhalf = wave & 1;
  const int blk = blockIdx.x;

  const int kb = (lane < 16) ? 0 : 8;     // B fragment K offset per lane half
  const int kbase = (lane < 16) ? 0 : 8;  // A fragment K offset per lane half
  const int m0 = lane & 15;
  const unsigned ldsbase = (unsigned)(size_t)(&xh[0][0]);

  unsigned tgt = 0;

  #pragma unroll 1
  for (int l = 0; l < NLAYER; ++l) {
    const __bf16* inp = (l == 0) ? xb : hall;                 // layer1 reads layer0's h sequence
    __bf16* hl = hall + (size_t)l * SEQ * BATCH * HID;
    const __bf16* Wl = Wcat + (size_t)l * GATES * KDIM;
    const float* bl = biasAll + (size_t)l * GATES;
    float* cl = cbAll + (size_t)l * BATCH * HID;
    float* outp = (l == NLAYER - 1) ? out : nullptr;
    float* hT = hTall + (size_t)l * BATCH * HID;
    float* cT = cTall + (size_t)l * BATCH * HID;

    // B-matrix: column n of B(KxN) == row n of Wcat[4H][2H] (contiguous K)
    const int n = gate * HID + blk * 32 + jhalf * 16 + (lane & 15);
    const __bf16* Wrow = Wl + (size_t)n * KDIM;

    #pragma unroll 1
    for (int t = 0; t < SEQ; ++t) {
      const __bf16* xt = inp + (size_t)t * BATCH * HID;
      const __bf16* hprev = (t == 0) ? (h0b + (size_t)l * BATCH * HID)
                                     : (hl + (size_t)(t - 1) * BATCH * HID);

      v8f acc0 = {};  // M rows 0..15
      v8f acc1 = {};  // M rows 16..31

      #pragma unroll 1
      for (int kb0 = 0; kb0 < KDIM; kb0 += KCHUNK) {
        // ---- stage 32x512 bf16 activation chunk into LDS via Tensor Data Mover ----
        const __bf16* src = (kb0 < HID) ? (xt + kb0) : (hprev + (kb0 - HID));
        if (wave == 0) {
          unsigned long long ga = (unsigned long long)(size_t)src;
          u32x4 g0;
          g0[0] = 1u;                                       // count = 1
          g0[1] = ldsbase;                                  // LDS byte address
          g0[2] = (unsigned)(ga & 0xffffffffu);             // global_addr[31:0]
          g0[3] = (unsigned)((ga >> 32) & 0x1ffffffu) | (2u << 30);  // addr[56:32] | type=2
          i32x8 g1;
          g1[0] = (int)((1u << 16)     // data_size = 2 bytes
                      | (1u << 20)     // pad_enable
                      | (7u << 22)     // pad_interval = 256 DWORDs (one 512-elem row)
                      | (3u << 25));   // pad_amount = 4 DWORDs (8 bf16)
          g1[1] = (int)((KCHUNK & 0xffffu) << 16);          // tensor_dim0 = 512
          g1[2] = (int)((BATCH & 0xffffu) << 16);           // tensor_dim1 = 32
          g1[3] = (int)((KCHUNK & 0xffffu) << 16);          // tile_dim0 = 512
          g1[4] = (int)(BATCH & 0xffffu);                   // tile_dim1 = 32
          g1[5] = (int)HID;                                 // tensor_dim0_stride = 1024
          g1[6] = 0;
          g1[7] = 0;
          i32x4 gz = {0, 0, 0, 0};
          i32x8 gz8 = {0, 0, 0, 0, 0, 0, 0, 0};
          __builtin_amdgcn_tensor_load_to_lds(g0, g1, gz, gz, gz8, 0);
          __builtin_amdgcn_s_wait_tensorcnt(0);
        }
        __syncthreads();

        // WGP-scope prefetch of next K-chunk of the weight stream
        if (kb0 + KCHUNK < KDIM) __builtin_prefetch(Wrow + kb0 + KCHUNK + kb, 0, 3);

        #pragma unroll 8
        for (int kc = 0; kc < KCHUNK / 32; ++kc) {
          const int kl = kc * 32;
          V16 a0, a1, b;
          const u32x4* ap0 = (const u32x4*)&xh[m0][kl + kbase];
          a0.u[0] = ap0[0]; a0.u[1] = ap0[2];
          const u32x4* ap1 = (const u32x4*)&xh[m0 + 16][kl + kbase];
          a1.u[0] = ap1[0]; a1.u[1] = ap1[2];
          const u32x4* bp = (const u32x4*)(Wrow + kb0 + kl + kb);
          b.u[0] = bp[0]; b.u[1] = bp[2];

          acc0 = __builtin_amdgcn_wmma_f32_16x16x32_bf16(false, a0.v, false, b.v, (short)0, acc0, false, false);
          acc1 = __builtin_amdgcn_wmma_f32_16x16x32_bf16(false, a1.v, false, b.v, (short)0, acc1, false, false);
        }
        __syncthreads();
      }

      // scatter accumulators into gate buffer (C layout: lanes<16 -> M=r, lanes>=16 -> M=8+r)
      {
        const int jl = jhalf * 16 + (lane & 15);
        const int mb = (lane < 16) ? 0 : 8;
        #pragma unroll
        for (int r = 0; r < 8; ++r) {
          gbuf[gate][mb + r][jl] = acc0[r];
          gbuf[gate][16 + mb + r][jl] = acc1[r];
        }
      }
      __syncthreads();

      // elementwise LSTM cell on this WG's 32 batch x 32 columns
      #pragma unroll
      for (int p = 0; p < 4; ++p) {
        int idx = tid + p * 256;
        int b_ = idx >> 5;
        int jl = idx & 31;
        int j = blk * 32 + jl;
        float ig = gbuf[0][b_][jl] + bl[0 * HID + j];
        float fg = gbuf[1][b_][jl] + bl[1 * HID + j];
        float gg = gbuf[2][b_][jl] + bl[2 * HID + j];
        float og = gbuf[3][b_][jl] + bl[3 * HID + j];
        float cold = cl[b_ * HID + j];
        float cnew = sigmoidf_(fg) * cold + sigmoidf_(ig) * tanhf(gg);
        float hnew = sigmoidf_(og) * tanhf(cnew);
        cl[b_ * HID + j] = cnew;
        hl[(size_t)t * BATCH * HID + b_ * HID + j] = (__bf16)hnew;   // recurrent / next-layer input
        if (outp) outp[(size_t)b_ * SEQ * HID + (size_t)t * HID + j] = hnew;
        if (t == SEQ - 1) {
          hT[b_ * HID + j] = hnew;
          cT[b_ * HID + j] = cnew;
        }
      }

      // device-wide barrier before next timestep consumes h
      tgt += NBLK;
      grid_sync(bar, tgt);
    }
  }
}

// ---------------- host orchestration ----------------

extern "C" void kernel_launch(void* const* d_in, const int* in_sizes, int n_in,
                              void* d_out, int out_size, void* d_ws, size_t ws_size,
                              hipStream_t stream) {
  (void)in_sizes; (void)n_in; (void)out_size; (void)ws_size;
  const float* x  = (const float*)d_in[0];
  const float* h0 = (const float*)d_in[1];
  const float* c0 = (const float*)d_in[2];
  const float* Wi = (const float*)d_in[3];
  const float* Wh = (const float*)d_in[4];
  const float* bi = (const float*)d_in[5];
  const float* bh = (const float*)d_in[6];

  char* p = (char*)d_ws;
  auto alloc = [&](size_t bytes) -> char* {
    char* r = p;
    p += (bytes + 255) & ~(size_t)255;
    return r;
  };
  __bf16* Wcat = (__bf16*)alloc((size_t)NLAYER * GATES * KDIM * 2);
  float*  bias = (float*) alloc((size_t)NLAYER * GATES * 4);
  __bf16* xb   = (__bf16*)alloc((size_t)SEQ * BATCH * HID * 2);
  __bf16* hall = (__bf16*)alloc((size_t)NLAYER * SEQ * BATCH * HID * 2);
  __bf16* h0b  = (__bf16*)alloc((size_t)NLAYER * BATCH * HID * 2);
  float*  cb   = (float*) alloc((size_t)NLAYER * BATCH * HID * 4);
  unsigned* bar = (unsigned*)alloc(256);

  float* out = (float*)d_out;                       // [B][S][H]
  float* hT = out + (size_t)BATCH * SEQ * HID;      // [L][B][H]
  float* cT = hT + (size_t)NLAYER * BATCH * HID;    // [L][B][H]

  {
    int total = NLAYER * GATES * KDIM;
    prep_w<<<(total + 255) / 256, 256, 0, stream>>>(Wi, Wh, bi, bh, Wcat, bias, total);
  }
  {
    int total = SEQ * BATCH * HID;
    prep_x<<<(total + 255) / 256, 256, 0, stream>>>(x, xb, total);
  }
  {
    int total = NLAYER * BATCH * HID;
    prep_state<<<(total + 255) / 256, 256, 0, stream>>>(h0, c0, h0b, cb, bar, total);
  }

  lstm_persistent<<<NBLK, 256, 0, stream>>>(Wcat, bias, xb, hall, h0b, cb, out, hT, cT, bar);
}
